// GHLoss_19335942766650
// MI455X (gfx1250) — matrix operand order repaired and checked by reading, not compiled, not run on hardware
//
#include <hip/hip_runtime.h>
#include <hip/hip_bf16.h>

#define N_ROWS 16384
#define K_PROTO 4096
#define INV_L 20.0f            // 1 / 0.05
#define INV_G 10.0f            // 1 / 0.1
#define INV_N (1.0f / 16384.0f)
#define INV_K (1.0f / 4096.0f)

typedef __bf16 bf16_t;
typedef __bf16 v16bf __attribute__((ext_vector_type(16)));
typedef __bf16 v8bf  __attribute__((ext_vector_type(8)));
typedef float  v8f   __attribute__((ext_vector_type(8)));
typedef float  v4f   __attribute__((ext_vector_type(4)));

// ---------------------------------------------------------------------------
// Pass 0: per-row log-sum-exp of x/gamma (stable), optionally cache
// E[n,k] = bf16(exp(x[n,k]/L)) for L2-resident Sinkhorn iterations.
// One block (256 threads) per row; row kept in registers (16 f32/thread).
// ---------------------------------------------------------------------------
template<bool CACHED>
__global__ void k_prep(const float* __restrict__ X, float* __restrict__ lse,
                       bf16_t* __restrict__ E) {
  __shared__ float red[256];
  const int n = blockIdx.x;
  const int t = threadIdx.x;
  const float* row = X + (size_t)n * K_PROTO;
  float x[16];
  float m = -3.4e38f;
#pragma unroll
  for (int j = 0; j < 16; ++j) {
    x[j] = row[t + j * 256];
    m = fmaxf(m, x[j] * INV_G);
  }
  red[t] = m; __syncthreads();
  for (int s = 128; s > 0; s >>= 1) {
    if (t < s) red[t] = fmaxf(red[t], red[t + s]);
    __syncthreads();
  }
  m = red[0]; __syncthreads();
  float ssum = 0.0f;
#pragma unroll
  for (int j = 0; j < 16; ++j) ssum += __expf(x[j] * INV_G - m);
  red[t] = ssum; __syncthreads();
  for (int s = 128; s > 0; s >>= 1) {
    if (t < s) red[t] += red[t + s];
    __syncthreads();
  }
  if (t == 0) lse[n] = m + __logf(red[0]);
  if (CACHED) {
    bf16_t* er = E + (size_t)n * K_PROTO;
#pragma unroll
    for (int j = 0; j < 16; ++j) er[t + j * 256] = (bf16_t)__expf(x[j] * INV_L);
  }
}

// ---------------------------------------------------------------------------
// colacc[k] += sum_n e[n,k] * c[n]     (c == nullptr -> c = 1/N, iteration 0)
// Thread <-> column: perfectly coalesced streaming; grid.y splits rows.
// ---------------------------------------------------------------------------
template<bool CACHED>
__global__ void k_colsum(const void* __restrict__ Xv, const float* __restrict__ cvec,
                         float* __restrict__ colacc, int rows_per_blk) {
  const int col = blockIdx.x * 256 + threadIdx.x;
  const int r0 = blockIdx.y * rows_per_blk;
  float acc = 0.0f;
  if (CACHED) {
    const bf16_t* Ep = (const bf16_t*)Xv;
    for (int r = r0; r < r0 + rows_per_blk; ++r) {
      float e = (float)Ep[(size_t)r * K_PROTO + col];
      float cv = cvec ? cvec[r] : INV_N;
      acc = fmaf(e, cv, acc);
    }
  } else {
    const float* Xp = (const float*)Xv;
    for (int r = r0; r < r0 + rows_per_blk; ++r) {
      float e = __expf(Xp[(size_t)r * K_PROTO + col] * INV_L);
      float cv = cvec ? cvec[r] : INV_N;
      acc = fmaf(e, cv, acc);
    }
  }
  atomicAdd(&colacc[col], acc);
}

// r[k] = (1/K) / colacc[k]   (f32 for the loss pass, bf16 for the WMMA GEMV)
__global__ void k_make_r(const float* __restrict__ colacc, float* __restrict__ rf,
                         bf16_t* __restrict__ rb) {
  int k = blockIdx.x * 256 + threadIdx.x;
  float r = INV_K / colacc[k];
  rf[k] = r;
  rb[k] = (bf16_t)r;
}

// ---------------------------------------------------------------------------
// c_out[n] = (1/N) / sum_k e[n,k]*r[k] via v_wmma_f32_16x16x32_bf16.
// One wave handles 16 rows x full K. ISA layouts (CDNA5 ISA):
//  A 16x32 bf16: lane L(<16)=row L, elems 0-7 -> K h*8+0..7, 8-15 -> K 16+h*8+..
//  B 32x16 bf16: same K order as A, elems 0-7 -> K h*8+0..7, 8-15 -> K 16+h*8+.. (r replicated
//  across all 16 B columns, so every D column is the same rowsum).
//  D: vgpr v, lanes 0-15 -> M=v ; lanes 16-31 -> M=v+8.
// ---------------------------------------------------------------------------
template<bool CACHED>
__global__ void k_rowsum_wmma(const void* __restrict__ Xv,
                              const bf16_t* __restrict__ rb,
                              float* __restrict__ cout) {
  const int tid  = threadIdx.x;
  const int wave = tid >> 5;
  const int lane = tid & 31;
  const int hh   = lane >> 4;   // half of the wave
  const int l16  = lane & 15;
  const int n0   = (blockIdx.x * 8 + wave) * 16;
  const int arow = n0 + l16;    // A-matrix row this lane feeds (both halves)

  v8f acc = {};
  for (int k0 = 0; k0 < K_PROTO; k0 += 32) {
    v16bf b;
    {
      const bf16_t* rp = rb + k0 + hh * 8;
      v8bf b0 = *(const v8bf*)(rp);
      v8bf b1 = *(const v8bf*)(rp + 16);
#pragma unroll
      for (int i = 0; i < 8; ++i) { b[i] = b0[i]; b[8 + i] = b1[i]; }
    }
    v16bf a;
    if constexpr (CACHED) {
      const bf16_t* row = (const bf16_t*)Xv + (size_t)arow * K_PROTO;
      v8bf a0 = *(const v8bf*)(row + k0 + hh * 8);
      v8bf a1 = *(const v8bf*)(row + k0 + 16 + hh * 8);
#pragma unroll
      for (int i = 0; i < 8; ++i) { a[i] = a0[i]; a[8 + i] = a1[i]; }
    } else {
      const float* row = (const float*)Xv + (size_t)arow * K_PROTO;
      v4f f0 = *(const v4f*)(row + k0 + hh * 8);
      v4f f1 = *(const v4f*)(row + k0 + hh * 8 + 4);
      v4f f2 = *(const v4f*)(row + k0 + 16 + hh * 8);
      v4f f3 = *(const v4f*)(row + k0 + 16 + hh * 8 + 4);
#pragma unroll
      for (int i = 0; i < 4; ++i) {
        a[i]      = (bf16_t)__expf(f0[i] * INV_L);
        a[4 + i]  = (bf16_t)__expf(f1[i] * INV_L);
        a[8 + i]  = (bf16_t)__expf(f2[i] * INV_L);
        a[12 + i] = (bf16_t)__expf(f3[i] * INV_L);
      }
    }
    acc = __builtin_amdgcn_wmma_f32_16x16x32_bf16(false, a, false, b,
                                                  (short)0, acc, false, false);
  }
  if (l16 == 0) {
    const int base = n0 + hh * 8;   // lane 0 -> rows n0..n0+7, lane 16 -> +8
#pragma unroll
    for (int v = 0; v < 8; ++v) cout[base + v] = INV_N / acc[v];
  }
}

// ---------------------------------------------------------------------------
// Final fused loss pass. One block per row; r vectors staged in LDS.
// loss -= c_x[n]*(s1x - lse_y[n]*s0x) + c_y[n]*(s1y - lse_x[n]*s0y)
// where s0x = sum_k r_x[k] e1[n,k], s1x = sum_k r_x[k] e1[n,k] * x2[n,k]/g.
// ---------------------------------------------------------------------------
__global__ void k_loss(const float* __restrict__ X1, const float* __restrict__ X2,
                       const float* __restrict__ rx, const float* __restrict__ ry,
                       const float* __restrict__ cx, const float* __restrict__ cy,
                       const float* __restrict__ lsex, const float* __restrict__ lsey,
                       float* __restrict__ out) {
  __shared__ float s_rx[K_PROTO];
  __shared__ float s_ry[K_PROTO];
  __shared__ float red[4][256];
  const int n = blockIdx.x, t = threadIdx.x;
#pragma unroll
  for (int j = 0; j < 16; ++j) {
    int k = t + j * 256;
    s_rx[k] = rx[k];
    s_ry[k] = ry[k];
  }
  __syncthreads();
  const float* r1 = X1 + (size_t)n * K_PROTO;
  const float* r2 = X2 + (size_t)n * K_PROTO;
  float s0x = 0.f, s1x = 0.f, s0y = 0.f, s1y = 0.f;
#pragma unroll
  for (int j = 0; j < 16; ++j) {
    int k = t + j * 256;
    float x1 = r1[k], x2 = r2[k];
    float t1 = s_rx[k] * __expf(x1 * INV_L);
    float t2 = s_ry[k] * __expf(x2 * INV_L);
    s0x += t1; s1x = fmaf(t1, x2 * INV_G, s1x);
    s0y += t2; s1y = fmaf(t2, x1 * INV_G, s1y);
  }
  red[0][t] = s0x; red[1][t] = s1x; red[2][t] = s0y; red[3][t] = s1y;
  __syncthreads();
  for (int s = 128; s > 0; s >>= 1) {
    if (t < s) {
#pragma unroll
      for (int q = 0; q < 4; ++q) red[q][t] += red[q][t + s];
    }
    __syncthreads();
  }
  if (t == 0) {
    float termx = cx[n] * (red[1][0] - lsey[n] * red[0][0]);
    float termy = cy[n] * (red[3][0] - lsex[n] * red[2][0]);
    atomicAdd(out, -(termx + termy));   // N * c[n] * (1/N) == c[n]
  }
}

// ---------------------------------------------------------------------------
extern "C" void kernel_launch(void* const* d_in, const int* in_sizes, int n_in,
                              void* d_out, int out_size, void* d_ws, size_t ws_size,
                              hipStream_t stream) {
  (void)in_sizes; (void)n_in; (void)out_size;
  const float* c1 = (const float*)d_in[0];
  const float* c2 = (const float*)d_in[1];

  char* const base = (char*)d_ws;
  char* cur = base;
  auto carve = [&](size_t b) -> void* {
    void* p = (void*)cur;
    cur += (b + 255) & ~(size_t)255;
    return p;
  };
  float*  colacc = (float*)carve(K_PROTO * sizeof(float));
  float*  r_x    = (float*)carve(K_PROTO * sizeof(float));
  float*  r_y    = (float*)carve(K_PROTO * sizeof(float));
  bf16_t* r16    = (bf16_t*)carve(K_PROTO * sizeof(bf16_t));
  float*  c_x    = (float*)carve(N_ROWS * sizeof(float));
  float*  c_y    = (float*)carve(N_ROWS * sizeof(float));
  float*  lse_x  = (float*)carve(N_ROWS * sizeof(float));
  float*  lse_y  = (float*)carve(N_ROWS * sizeof(float));

  const size_t Eb = (size_t)N_ROWS * K_PROTO * sizeof(bf16_t);  // 128 MB
  const bool cached = ws_size >= (size_t)(cur - base) + 2 * Eb; // deterministic in ws_size
  bf16_t* E1 = cached ? (bf16_t*)carve(Eb) : nullptr;
  bf16_t* E2 = cached ? (bf16_t*)carve(Eb) : nullptr;

  const dim3 csGrid(K_PROTO / 256, 64);
  const int rowsPerBlk = N_ROWS / 64;

  auto run_sinkhorn = [&](const float* X, bf16_t* E, float* lse, float* rf, float* cvec) {
    if (E) k_prep<true ><<<N_ROWS, 256, 0, stream>>>(X, lse, E);
    else   k_prep<false><<<N_ROWS, 256, 0, stream>>>(X, lse, nullptr);
    const float* cv = nullptr;
    for (int it = 0; it < 3; ++it) {
      hipMemsetAsync(colacc, 0, K_PROTO * sizeof(float), stream);
      if (E) k_colsum<true ><<<csGrid, 256, 0, stream>>>(E, cv, colacc, rowsPerBlk);
      else   k_colsum<false><<<csGrid, 256, 0, stream>>>(X, cv, colacc, rowsPerBlk);
      k_make_r<<<K_PROTO / 256, 256, 0, stream>>>(colacc, rf, r16);
      if (E) k_rowsum_wmma<true ><<<N_ROWS / 128, 256, 0, stream>>>(E, r16, cvec);
      else   k_rowsum_wmma<false><<<N_ROWS / 128, 256, 0, stream>>>(X, r16, cvec);
      cv = cvec;
    }
  };

  run_sinkhorn(c1, E1, lse_x, r_x, c_x);
  run_sinkhorn(c2, E2, lse_y, r_y, c_y);

  hipMemsetAsync(d_out, 0, sizeof(float), stream);
  k_loss<<<N_ROWS, 256, 0, stream>>>(c1, c2, r_x, r_y, c_x, c_y,
                                     lse_x, lse_y, (float*)d_out);
}
